// NeuralField_18605798326294
// MI455X (gfx1250) — compile-verified, matrix operand order checked by reading, not run on hardware
//
#include <hip/hip_runtime.h>
#include <hip/hip_bf16.h>

// ---------------------------------------------------------------------------
// Fused NeRF-MLP inference for gfx1250 (MI455X), wave32 + WMMA f16->f32.
// One wave = 32 points (two 16-row M-tiles): every weight B-fragment load
// feeds two WMMAs. B fragments are software-pipelined (depth-2 rotation) and
// A fragments (LDS) prefetched one k-step ahead so the steady state keeps
// loads in flight instead of s_wait_loadcnt 0 round-trips per WMMA pair.
// Activations staged in LDS as [hid(128) | enc(96)] rows, stride 232 halfs.
// Weights pre-converted to f16 and pre-swizzled into WMMA B-fragment lane
// order in d_ws by a prep kernel.
// ---------------------------------------------------------------------------

typedef __attribute__((ext_vector_type(16))) _Float16 v16h;
typedef __attribute__((ext_vector_type(8)))  _Float16 v8h;
typedef __attribute__((ext_vector_type(8)))  float    v8f;

#define NPTS        524288
#define DIM_ENC     96
#define DIM_HID     128
#define DIM_IN_HID  224
#define DIM_OUT     4
#define NHID        7          // structurally distinct hidden layers
#define ROWS        232        // LDS row stride in halfs: 128 hid + 96 enc + 8 pad
#define WAVES       4          // waves per workgroup (128 threads)
#define PTS_PER_WAVE 32        // two 16-point M-tiles
#define FRAG_HALFS  512        // one B fragment: 32 lanes x 16 halfs
// fragment counts / offsets (in fragments)
#define F_W0        24         // 3 k-steps * 8 n-chunks
#define F_WH        56         // 7 k-steps * 8 n-chunks (per hidden layer)
#define F_WL        7          // 7 k-steps * 1 n-chunk
#define FRAGS_TOTAL (F_W0 + NHID*F_WH + F_WL)          // 423
#define TOTAL_HALFS (FRAGS_TOTAL * FRAG_HALFS)          // 216576 (433,152 B in d_ws)

// ---------------------------------------------------------------------------
// Prep: convert f32 weights -> f16 B-fragments, pre-swizzled so that the main
// kernel's per-lane fragment load is a single contiguous 32-byte read.
// B-fragment layout (mirror of the documented 16-bit A layout):
//   lane L: column n = L%16, group g = L/16
//   half element e: K = ks*32 + g*8 + (e&7) + (e>=8 ? 16 : 0)
// ---------------------------------------------------------------------------
__global__ __launch_bounds__(256) void nf_prep(const float* __restrict__ W0,
                                               const float* __restrict__ Wh,
                                               const float* __restrict__ Wl,
                                               _Float16* __restrict__ wf) {
  int idx = blockIdx.x * 256 + threadIdx.x;
  if (idx >= TOTAL_HALFS) return;
  int s    = idx & (FRAG_HALFS - 1);   // position within fragment
  int f    = idx >> 9;                 // global fragment index
  int lane = s >> 4;
  int e    = s & 15;
  int gg   = lane >> 4;
  int nn   = lane & 15;
  int krow = (gg << 3) + (e & 7) + ((e >> 3) << 4);  // K within 32-wide block

  const float* W;
  int outdim, ks, nc;
  if (f < F_W0) {                       // input layer: 96 -> 128
    W = W0; outdim = DIM_HID; ks = f >> 3; nc = f & 7;
  } else if (f < F_W0 + NHID * F_WH) {  // hidden layers: 224 -> 128
    int h = f - F_W0;
    int l = h / F_WH;
    int r = h % F_WH;
    W = Wh + l * DIM_IN_HID * DIM_HID;
    outdim = DIM_HID; ks = r >> 3; nc = r & 7;
  } else {                              // output layer: 224 -> 4 (zero padded)
    int r = f - (F_W0 + NHID * F_WH);
    W = Wl; outdim = DIM_OUT; ks = r; nc = 0;
  }
  int K = ks * 32 + krow;
  int n = nc * 16 + nn;
  float v = (n < outdim) ? W[K * outdim + n] : 0.0f;
  wf[idx] = (_Float16)v;
}

// ---------------------------------------------------------------------------
// Main fused kernel helpers
// ---------------------------------------------------------------------------
__device__ __forceinline__ v16h load_a(const _Float16* rowp, int kb) {
  // A fragment, 16-bit 16x32: this lane's halfs kb..kb+7 and kb+16..kb+23
  v8h lo = *(const v8h*)(rowp + kb);
  v8h hi = *(const v8h*)(rowp + kb + 16);
  return __builtin_shufflevector(lo, hi, 0, 1, 2, 3, 4, 5, 6, 7,
                                 8, 9, 10, 11, 12, 13, 14, 15);
}

__device__ __forceinline__ v8f wmma16(v16h a, v16h b, v8f c) {
  return __builtin_amdgcn_wmma_f32_16x16x32_f16(
      /*neg_a=*/false, a, /*neg_b=*/false, b,
      /*c_mod=*/(short)0, c, /*reuse_a=*/false, /*reuse_b=*/false);
}

// One dense layer (in -> 128, ReLU) over TWO 16-point tiles, software
// pipelined. KSTEPS k-blocks of 32 starting at half offset KOFF within the
// activation row. All 16 accumulators are produced before any writeback, so
// hidden-region reads/writes don't race.
template <int KSTEPS, int KOFF>
__device__ __forceinline__ void dense_layer2(_Float16* base,
                                             const _Float16* row0,
                                             const _Float16* row1,
                                             const v16h* __restrict__ wf,
                                             const float* __restrict__ bias,
                                             int g, int m, int lane) {
  constexpr int NF = KSTEPS * 8;
  v8f acc0[8], acc1[8];
#pragma unroll
  for (int nc = 0; nc < 8; ++nc) {
    acc0[nc] = v8f{0, 0, 0, 0, 0, 0, 0, 0};
    acc1[nc] = v8f{0, 0, 0, 0, 0, 0, 0, 0};
  }

  // hoist bias loads so writeback doesn't stall on fresh global loads
  float bv[8];
#pragma unroll
  for (int nc = 0; nc < 8; ++nc) bv[nc] = bias[nc * 16 + m];

  // prime the pipeline: A for ks=0, B for f=0 and f=1
  v16h a0 = load_a(row0, KOFF + g * 8);
  v16h a1 = load_a(row1, KOFF + g * 8);
  v16h b0 = wf[lane];
  v16h b1 = wf[32 + lane];

#pragma unroll
  for (int f = 0; f < NF; ++f) {
    const int ks = f >> 3;
    const int nc = f & 7;
    // B prefetch, depth 2
    v16h bn = b1;
    if (f + 2 < NF) bn = wf[(f + 2) * 32 + lane];
    // A prefetch, one k-step ahead (last chunk of each k-step)
    v16h a0n = a0, a1n = a1;
    if (nc == 7 && ks + 1 < KSTEPS) {
      a0n = load_a(row0, KOFF + (ks + 1) * 32 + g * 8);
      a1n = load_a(row1, KOFF + (ks + 1) * 32 + g * 8);
    }
    acc0[nc] = wmma16(a0, b0, acc0[nc]);
    acc1[nc] = wmma16(a1, b0, acc1[nc]);
    b0 = b1; b1 = bn;
    a0 = a0n; a1 = a1n;
  }

  // D layout: lane -> column n = m (g picks rows 8..15); VGPR r -> row.
#pragma unroll
  for (int nc = 0; nc < 8; ++nc) {
#pragma unroll
    for (int r = 0; r < 8; ++r) {
      float v0 = acc0[nc][r] + bv[nc];
      float v1 = acc1[nc][r] + bv[nc];
      v0 = v0 > 0.0f ? v0 : 0.0f;
      v1 = v1 > 0.0f ? v1 : 0.0f;
      base[(r + g * 8) * ROWS + nc * 16 + m]      = (_Float16)v0;
      base[(16 + r + g * 8) * ROWS + nc * 16 + m] = (_Float16)v1;
    }
  }
}

// ---------------------------------------------------------------------------
// Main kernel: 128 threads = 4 waves, 32 points per wave, grid = N/128.
// ---------------------------------------------------------------------------
__global__ __launch_bounds__(128) void nf_main(const float* __restrict__ qp,
                                               const float* __restrict__ b0,
                                               const float* __restrict__ bh,
                                               const float* __restrict__ bl,
                                               const _Float16* __restrict__ wfrag,
                                               float* __restrict__ out) {
  __shared__ __align__(16) _Float16 smem[WAVES * PTS_PER_WAVE * ROWS];  // 59,392 B
  const int tid  = threadIdx.x;
  const int w    = tid >> 5;
  const int lane = tid & 31;
  const int g    = lane >> 4;
  const int m    = lane & 15;
  _Float16* base = smem + w * (PTS_PER_WAVE * ROWS);
  _Float16* row0 = base + m * ROWS;                // tile 0, this lane's point
  _Float16* row1 = base + (16 + m) * ROWS;         // tile 1, this lane's point
  const int p0 = (blockIdx.x * WAVES + w) * PTS_PER_WAVE;

  // --- positional encoding into enc region (row offset 128) -----------------
  // lane pair (g=0,1) with same m covers freqs 0..7 / 8..15 of both points.
  const float PI = 3.14159265358979323846f;
#pragma unroll
  for (int t = 0; t < 2; ++t) {
    _Float16* rowp = t ? row1 : row0;
    int p = p0 + t * 16 + m;
    float coords[3];
    coords[0] = qp[p * 3 + 0];
    coords[1] = qp[p * 3 + 1];
    coords[2] = qp[p * 3 + 2];
#pragma unroll
    for (int c = 0; c < 3; ++c) {
      float sc = PI * (g ? 256.0f : 1.0f);  // pi * 2^(g*8)
#pragma unroll
      for (int u = 0; u < 8; ++u) {
        float arg = coords[c] * sc;
        int j = 128 + c * 32 + g * 8 + u;   // sin slot; cos at +16
        rowp[j]      = (_Float16)__sinf(arg);
        rowp[j + 16] = (_Float16)__cosf(arg);
        sc *= 2.0f;
      }
    }
  }
  // All LDS traffic below is same-wave (in-order per ISA); no barrier needed.

  const v16h* wfv = (const v16h*)wfrag;

  // layer 0: enc(96) -> hid(128), reads from enc region (KOFF=128)
  dense_layer2<3, 128>(base, row0, row1, wfv, b0, g, m, lane);

  // 7 hidden layers: [hid|enc](224) -> hid(128)
#pragma unroll 1
  for (int l = 0; l < NHID; ++l) {
    dense_layer2<7, 0>(base, row0, row1, wfv + (F_W0 + l * F_WH) * 32,
                       bh + l * DIM_HID, g, m, lane);
  }

  // output layer: 224 -> 4 (padded to 16 cols), no ReLU, both tiles,
  // same software pipelining.
  v8f acc0 = v8f{0, 0, 0, 0, 0, 0, 0, 0};
  v8f acc1 = v8f{0, 0, 0, 0, 0, 0, 0, 0};
  const v16h* wl = wfv + (F_W0 + NHID * F_WH) * 32;
  {
    v16h a0 = load_a(row0, g * 8);
    v16h a1 = load_a(row1, g * 8);
    v16h b0v = wl[lane];
    v16h b1v = wl[32 + lane];
#pragma unroll
    for (int ks = 0; ks < 7; ++ks) {
      v16h bn = b1v;
      if (ks + 2 < 7) bn = wl[(ks + 2) * 32 + lane];
      v16h a0n = a0, a1n = a1;
      if (ks + 1 < 7) {
        a0n = load_a(row0, (ks + 1) * 32 + g * 8);
        a1n = load_a(row1, (ks + 1) * 32 + g * 8);
      }
      acc0 = wmma16(a0, b0v, acc0);
      acc1 = wmma16(a1, b0v, acc1);
      b0v = b1v; b1v = bn;
      a0 = a0n; a1 = a1n;
    }
  }
  if (m < DIM_OUT) {
    float bv = bl[m];
#pragma unroll
    for (int r = 0; r < 8; ++r) {
      out[(p0 + r + g * 8) * DIM_OUT + m]      = acc0[r] + bv;
      out[(p0 + 16 + r + g * 8) * DIM_OUT + m] = acc1[r] + bv;
    }
  }
}

// ---------------------------------------------------------------------------
extern "C" void kernel_launch(void* const* d_in, const int* in_sizes, int n_in,
                              void* d_out, int out_size, void* d_ws, size_t ws_size,
                              hipStream_t stream) {
  const float* qp = (const float*)d_in[0];
  const float* W0 = (const float*)d_in[1];
  const float* b0 = (const float*)d_in[2];
  const float* Wh = (const float*)d_in[3];
  const float* bh = (const float*)d_in[4];
  const float* Wl = (const float*)d_in[5];
  const float* bl = (const float*)d_in[6];
  _Float16* wf = (_Float16*)d_ws;   // needs TOTAL_HALFS*2 = 433,152 bytes
  float* out = (float*)d_out;

  // weight convert + fragment swizzle (cheap, deterministic, every call)
  int prep_blocks = (TOTAL_HALFS + 255) / 256;  // 846
  nf_prep<<<prep_blocks, 256, 0, stream>>>(W0, Wh, Wl, wf);

  // fused MLP: 4096 blocks * 128 threads = 4 waves * 32 pts -> N exactly
  int main_blocks = NPTS / (WAVES * PTS_PER_WAVE);   // 4096
  nf_main<<<main_blocks, 128, 0, stream>>>(qp, b0, bh, bl, wf, out);
}
